// SAM3D_25864293056938
// MI455X (gfx1250) — matrix operand order repaired and checked by reading, not compiled, not run on hardware
//
#include <hip/hip_runtime.h>
#include <math.h>

// ---------------------------------------------------------------------------
// SAM3D spatial attention for MI455X (gfx1250, wave32).
//   x: (2, 64, 16, 128, 128) f32,  w: (1, 2, 7, 7, 7) f32
// Pass 1: channel max/mean -> f16 map[b][ci][t][h][w]   (2 MB, L2-resident)
// Pass 2: 7x7x7 conv via v_wmma_f32_16x16x32_f16 + sigmoid -> attn f32 (2 MB)
// Pass 3: out = attn * x   (streaming, HBM-bound)
// ---------------------------------------------------------------------------

typedef __attribute__((ext_vector_type(16))) _Float16 v16h;
typedef __attribute__((ext_vector_type(8)))  float    v8f;
typedef __attribute__((ext_vector_type(4)))  _Float16 h4;

#define S_SP   (1u << 18)   // T*H*W = 16*128*128 = 262144
#define P_SP   (1u << 19)   // B * S_SP
#define N_ELEM (1u << 25)   // B*C*S_SP = 33,554,432

// ---------------- Pass 1: channel max + mean -> f16 map ----------------
__global__ __launch_bounds__(256)
void reduce_kernel(const float* __restrict__ x, _Float16* __restrict__ map) {
  unsigned gid = blockIdx.x * blockDim.x + threadIdx.x;   // 131072 threads
  unsigned p = gid << 2;                                  // 4 spatial pts/thread
  unsigned b = p >> 18;
  unsigned s = p & (S_SP - 1);
  const float4* xv = (const float4*)x;
  unsigned base = ((b << 24) | s) >> 2;                   // (b*64*S + s)/4
  const unsigned stride = S_SP >> 2;                      // channel stride in float4

  float4 v0 = xv[base];
  float mx0 = v0.x, mx1 = v0.y, mx2 = v0.z, mx3 = v0.w;
  float sm0 = v0.x, sm1 = v0.y, sm2 = v0.z, sm3 = v0.w;
#pragma unroll 8
  for (int c = 1; c < 64; ++c) {
    float4 u = xv[base + (unsigned)c * stride];
    mx0 = fmaxf(mx0, u.x); mx1 = fmaxf(mx1, u.y);
    mx2 = fmaxf(mx2, u.z); mx3 = fmaxf(mx3, u.w);
    sm0 += u.x; sm1 += u.y; sm2 += u.z; sm3 += u.w;
  }
  const float inv = 1.0f / 64.0f;
  h4 hmax = { (_Float16)mx0, (_Float16)mx1, (_Float16)mx2, (_Float16)mx3 };
  h4 havg = { (_Float16)(sm0 * inv), (_Float16)(sm1 * inv),
              (_Float16)(sm2 * inv), (_Float16)(sm3 * inv) };
  h4* mp = (h4*)map;
  unsigned omax = ((b * 2u) * S_SP + s) >> 2;   // map[b][0][s]
  mp[omax] = hmax;
  mp[omax + (S_SP >> 2)] = havg;                // map[b][1][s]
}

// ---------------- Weight repack: (ci,dz,dy) groups, dx padded to 8 ---------
// wpad[chunk*32 + k], chunk 0..24, k 0..31:  g = chunk*4 + k/8, dx = k%8
__global__ __launch_bounds__(256)
void wpad_kernel(const float* __restrict__ w, _Float16* __restrict__ wpad) {
  unsigned i = blockIdx.x * blockDim.x + threadIdx.x;     // 1024 threads
  if (i >= 1024) return;
  float val = 0.0f;
  if (i < 800) {
    unsigned chunk = i >> 5, k = i & 31;
    unsigned g = chunk * 4 + (k >> 3);
    unsigned dx = k & 7;
    if (g < 98 && dx < 7) {
      unsigned ci = g / 49, rem = g - ci * 49;
      unsigned dz = rem / 7, dy = rem - dz * 7;
      val = w[((ci * 7 + dz) * 7 + dy) * 7 + dx];
    }
  }
  wpad[i] = (_Float16)val;
}

// ---------------- Pass 2: conv via WMMA + sigmoid -> attn ----------------
// One wave computes 16 outputs along W. 25 accumulating v_wmma_f32_16x16x32_f16,
// B = weight chunk replicated over all 16 columns (single output channel).
__global__ __launch_bounds__(256)
void conv_wmma_kernel(const _Float16* __restrict__ map,
                      const _Float16* __restrict__ wpad,
                      float* __restrict__ attn) {
  __shared__ float smem[8 * 16];
  unsigned tid  = threadIdx.x;
  unsigned lane = tid & 31;
  unsigned wv   = tid >> 5;
  unsigned q    = blockIdx.x * 8u + wv;        // 32768 tiles total
  unsigned w0   = (q & 7u) << 4;               // W tile base
  unsigned rest = q >> 3;
  unsigned h    = rest & 127u; rest >>= 7;
  unsigned t    = rest & 15u;
  unsigned b    = rest >> 4;
  unsigned half = lane >> 4;                   // 0: lanes 0-15, 1: lanes 16-31
  unsigned m    = lane & 15u;                  // output index (A rows / lanes mod 16)

  v8f acc = {};
  for (int c = 0; c < 25; ++c) {
    // A fragment (16-bit 16x32 layout): lane holds K = half*8+{0..7} (run 0)
    // and K = 16+half*8+{0..7} (run 1)  ->  groups g = 4c+half and 4c+2+half.
    v16h a = {};
#pragma unroll
    for (int r = 0; r < 2; ++r) {
      unsigned g = (unsigned)(4 * c) + half + (unsigned)(2 * r);
      int rowbase = 0;
      bool ok = false;
      if (g < 98u) {
        unsigned ci = g / 49u, rem = g - ci * 49u;
        unsigned dz = rem / 7u, dy = rem - dz * 7u;
        int tz = (int)t + (int)dz - 3;
        int hy = (int)h + (int)dy - 3;
        ok = ((unsigned)tz < 16u) && ((unsigned)hy < 128u);
        rowbase = (int)((((b * 2u + ci) * 16u + (unsigned)tz) * 128u
                          + (unsigned)hy) * 128u);
      }
#pragma unroll
      for (int dx = 0; dx < 8; ++dx) {
        int wc = (int)(w0 + m) + dx - 3;
        _Float16 av = (_Float16)0.0f;
        if (ok && (unsigned)wc < 128u) av = map[rowbase + wc];
        a[r * 8 + dx] = av;
      }
    }
    // B fragment (32x16 f16): lane holds K = half*8+{0..7} and K = 16+half*8+{0..7},
    // identical for every column n -> two aligned 16B runs of wpad.
    v16h bm;
    const float4* wp = (const float4*)(wpad + (unsigned)c * 32u + half * 8u);
    ((float4*)&bm)[0] = wp[0];
    ((float4*)&bm)[1] = wp[2];

    acc = __builtin_amdgcn_wmma_f32_16x16x32_f16(
        /*neg_a=*/false, a, /*neg_b=*/false, bm,
        /*c_mod=*/(short)0, acc, /*reuse_a=*/false, /*reuse_b=*/false);
  }

  // C layout: lane n holds rows M=0..7 (acc[0..7]); lane n+16 holds M=8..15.
  // All columns identical -> column 0 (lanes 0 and 16) has the 16 results.
  if ((lane & 15u) == 0u) {
#pragma unroll
    for (int v = 0; v < 8; ++v) smem[wv * 16 + half * 8 + (unsigned)v] = acc[v];
  }
  __syncthreads();
  if (lane < 16u) {
    float val = smem[wv * 16 + lane];
    float sg = 1.0f / (1.0f + expf(-val));
    unsigned p = ((b * 16u + t) * 128u + h) * 128u + w0 + lane; // b*S + s
    attn[p] = sg;
  }
}

// ---------------- Pass 3: out = attn * x (broadcast over C) ----------------
__global__ __launch_bounds__(256)
void apply_kernel(const float* __restrict__ x, const float* __restrict__ attn,
                  float* __restrict__ out) {
  unsigned gid = blockIdx.x * blockDim.x + threadIdx.x;   // 8,388,608 threads
  unsigned e_hi = gid >> 22;                              // b = (gid*4) >> 24
  unsigned s4 = gid & ((S_SP >> 2) - 1);                  // (e & (S-1)) / 4
  const float4* xv = (const float4*)x;
  const float4* av = (const float4*)attn;
  float4 xx = xv[gid];
  float4 aa = av[(e_hi << 16) | s4];                      // (b*S + s)/4
  float4 o;
  o.x = xx.x * aa.x; o.y = xx.y * aa.y;
  o.z = xx.z * aa.z; o.w = xx.w * aa.w;
  ((float4*)out)[gid] = o;
}

// ---------------------------------------------------------------------------
extern "C" void kernel_launch(void* const* d_in, const int* in_sizes, int n_in,
                              void* d_out, int out_size, void* d_ws, size_t ws_size,
                              hipStream_t stream) {
  const float* x = (const float*)d_in[0];
  const float* w = (const float*)d_in[1];
  float* out = (float*)d_out;

  char* ws = (char*)d_ws;
  _Float16* map  = (_Float16*)ws;                    // 2 MB f16 map[b][ci][s]
  float*    attn = (float*)(ws + (2u << 20));        // 2 MB f32 attn[b][s]
  _Float16* wpad = (_Float16*)(ws + (4u << 20));     // 2 KB padded weights

  reduce_kernel<<<dim3(512),   dim3(256), 0, stream>>>(x, map);
  wpad_kernel  <<<dim3(4),     dim3(256), 0, stream>>>(w, wpad);
  conv_wmma_kernel<<<dim3(4096), dim3(256), 0, stream>>>(map, wpad, attn);
  apply_kernel <<<dim3(32768), dim3(256), 0, stream>>>(x, attn, out);
}
